// up_res_block_76785425318546
// MI455X (gfx1250) — matrix operand order repaired and checked by reading, not compiled, not run on hardware
//
#include <hip/hip_runtime.h>
#include <hip/hip_bf16.h>

typedef __attribute__((ext_vector_type(8)))  float  v8f;
typedef __attribute__((ext_vector_type(16))) __bf16 v16bf;

#define ALPHA 0.2f

// ---------- helpers ----------

__device__ __forceinline__ unsigned bf16_rne(float f) {
    unsigned u = __float_as_uint(f);
    return (u + 0x7fffu + ((u >> 16) & 1u)) >> 16;   // 16-bit bf16, round-nearest-even
}

// pack fp32 -> u32 with bits[15:0]=bf16_hi, bits[31:16]=bf16_lo (residual)
__device__ __forceinline__ unsigned pack_split(float f) {
    unsigned h = bf16_rne(f);
    float r = f - __uint_as_float(h << 16);
    unsigned l = bf16_rne(r);
    return h | (l << 16);
}

__device__ __forceinline__ uint4 pack_split4(float4 f) {
    uint4 r;
    r.x = pack_split(f.x);
    r.y = pack_split(f.y);
    r.z = pack_split(f.z);
    r.w = pack_split(f.w);
    return r;
}

__device__ __forceinline__ float leaky(float v) { return v >= 0.f ? v : ALPHA * v; }

__device__ __forceinline__ v8f wmma_bf16(v16bf a, v16bf b, v8f c) {
    return __builtin_amdgcn_wmma_f32_16x16x32_bf16(false, a, false, b, (short)0, c,
                                                   false, false);
}

// Build A hi/lo fragments (16-bit A 16x32 layout: lane = M%16 + K-half, dword j
// holds K pair base = (j<4 ? 2j : 8+2j) + 8*(lane/16)) from LDS of packed
// (hi|lo) u32 elements; base already includes cc0 + 8*kh.
__device__ __forceinline__ void build_afrags(const unsigned* __restrict__ p,
                                             v16bf& Ah, v16bf& Al) {
    union { unsigned u[8]; v16bf v; } h, l;
#pragma unroll
    for (int j = 0; j < 8; ++j) {
        int cb = (j < 4) ? (2 * j) : (8 + 2 * j);
        unsigned e0 = p[cb], e1 = p[cb + 1];
        // hi pair: {lo16(e0), lo16(e1)} ; lo pair: {hi16(e0), hi16(e1)}
        h.u[j] = __builtin_amdgcn_perm(e1, e0, 0x05040100u);
        l.u[j] = __builtin_amdgcn_perm(e1, e0, 0x07060302u);
    }
    Ah = h.v; Al = l.v;
}

__device__ __forceinline__ v16bf load_wfrag(const unsigned* __restrict__ p) {
    union { uint4 q[2]; v16bf v; } u;
    const uint4* q = (const uint4*)p;
    u.q[0] = q[0];
    u.q[1] = q[1];
    return u.v;
}

// ---------- prepass: fold BN ----------

__global__ void __launch_bounds__(256)
fold_bn(const float* __restrict__ b1, const float* __restrict__ g1,
        const float* __restrict__ be1, const float* __restrict__ m1,
        const float* __restrict__ v1,
        const float* __restrict__ b2, const float* __restrict__ g2,
        const float* __restrict__ be2, const float* __restrict__ m2,
        const float* __restrict__ v2,
        float* __restrict__ s1, float* __restrict__ s2,
        float* __restrict__ bias1, float* __restrict__ bias2) {
    int t = threadIdx.x;
    if (t < 256) {
        float s = g1[t] * rsqrtf(v1[t] + 1e-3f);
        s1[t] = s;
        bias1[t] = (b1[t] - m1[t]) * s + be1[t];
    }
    if (t < 64) {
        float s = g2[t] * rsqrtf(v2[t] + 1e-3f);
        s2[t] = s;
        bias2[t] = (b2[t] - m2[t]) * s + be2[t];
    }
}

// ---------- prepass: split weights to bf16 hi/lo in B-fragment order ----------
// dst dword index = ((it*nSubCnt + ns)*32 + lane)*8 + j
// B 32x16 bf16 layout: lane<16 -> n=lane, K=0..7,16..23 ; lane>=16 -> n=lane-16, K=8..15,24..31
// dword j <-> K pair (it*32 + 8*(lane/16) + (j<4 ? 2j : 8+2j), +1)

__global__ void __launch_bounds__(256)
pack_weights(const float* __restrict__ src, const float* __restrict__ scale,
             unsigned* __restrict__ hiDst, unsigned* __restrict__ loDst,
             int N, int nSubCnt, int total) {
    int tid = blockIdx.x * 256 + threadIdx.x;
    if (tid >= total) return;
    int j    = tid & 7;
    int lane = (tid >> 3) & 31;
    int ns   = (tid >> 8) % nSubCnt;
    int it   = tid / (256 * nSubCnt);
    int n = ns * 16 + (lane & 15);
    int k = it * 32 + 8 * (lane >> 4) + ((j < 4) ? (2 * j) : (8 + 2 * j));
    float s = scale ? scale[n] : 1.0f;
    float w0 = src[k * N + n] * s;
    float w1 = src[(k + 1) * N + n] * s;
    unsigned h0 = bf16_rne(w0);
    unsigned l0 = bf16_rne(w0 - __uint_as_float(h0 << 16));
    unsigned h1 = bf16_rne(w1);
    unsigned l1 = bf16_rne(w1 - __uint_as_float(h1 << 16));
    hiDst[tid] = h0 | (h1 << 16);
    loDst[tid] = l0 | (l1 << 16);
}

// ---------- stage 1: conv1(3x3,64->256) + bias/BN + leaky + subpixel -> T ----------
// Tile: 8x8 input pixels, all 256 output channels. 256 threads = 8 waves.
// Wave w: M-subtile = w%4 (16 pixels), N-half = w/4 (8 N-subtiles of 16).

__global__ void __launch_bounds__(256)
conv1_subpixel(const float* __restrict__ x,
               const unsigned* __restrict__ w1hi, const unsigned* __restrict__ w1lo,
               const float* __restrict__ bias1, unsigned* __restrict__ T) {
    __shared__ uint4 ldsX4[10 * 10 * 16];     // halo, packed (hi|lo) bf16 pairs
    const unsigned* ldsX = (const unsigned*)ldsX4;
    int bx = blockIdx.x;
    int px = bx & 15, py = (bx >> 4) & 15, b = bx >> 8;
    int x0 = px * 8, y0 = py * 8;
    int t = threadIdx.x;

    const float4* xv = (const float4*)x;
    for (int e = t; e < 1600; e += 256) {     // 16B-vectorized staging
        int c4 = e & 15, hc = (e >> 4) % 10, hr = e / 160;
        int yy = y0 + hr - 1, xx = x0 + hc - 1;
        float4 f = {0.f, 0.f, 0.f, 0.f};
        if (yy >= 0 && yy < 128 && xx >= 0 && xx < 128)
            f = xv[(((b * 128 + yy) * 128 + xx) << 4) + c4];
        ldsX4[e] = pack_split4(f);
    }
    __syncthreads();

    int lane = t & 31, w = t >> 5;
    int mSub = w & 3, nHalf = w >> 2;
    int kh = lane >> 4, ml = lane & 15;
    int pr = 2 * mSub + (ml >> 3), pc = ml & 7;   // A row = pixel (pr,pc) in 8x8

    v8f acc[8];
#pragma unroll
    for (int ns = 0; ns < 8; ++ns) acc[ns] = {};

    for (int it = 0; it < 18; ++it) {
        int blk = it >> 1;
        int ky = blk / 3, kx = blk % 3;
        int cc0 = (it & 1) << 5;
        v16bf Ah, Al;
        build_afrags(&ldsX[((pr + ky) * 10 + (pc + kx)) * 64 + cc0 + (kh << 3)], Ah, Al);
#pragma unroll
        for (int ns = 0; ns < 8; ++ns) {
            int nsg = nHalf * 8 + ns;
            const unsigned* ph = w1hi + (((it * 16 + nsg) * 32 + lane) << 3);
            const unsigned* pl = w1lo + (((it * 16 + nsg) * 32 + lane) << 3);
            v16bf Bh = load_wfrag(ph);
            v16bf Bl = load_wfrag(pl);
            acc[ns] = wmma_bf16(Ah, Bl, acc[ns]);
            acc[ns] = wmma_bf16(Al, Bh, acc[ns]);
            acc[ns] = wmma_bf16(Ah, Bh, acc[ns]);
        }
    }

    // epilogue: bias + leaky, bf16-split, subpixel scatter into T
#pragma unroll
    for (int ns = 0; ns < 8; ++ns) {
        int n = nHalf * 128 + ns * 16 + ml;      // conv1 out channel
        float bb = bias1[n];
        int g = n >> 2, hi2 = (n >> 1) & 1, wi = n & 1;
#pragma unroll
        for (int v = 0; v < 8; ++v) {
            int m = v + (kh << 3);               // row within 16-pixel subtile
            int p = 16 * mSub + m;
            int pr2 = p >> 3, pc2 = p & 7;
            int oy = 2 * (y0 + pr2) + wi, ox = 2 * (x0 + pc2) + hi2;
            float val = leaky(acc[ns][v] + bb);
            T[(((b * 256 + oy) * 256 + ox) << 6) + g] = pack_split(val);
        }
    }
}

// ---------- stage 2: conv2(3x3,64->64)+BN+leaky, + 1x1 skip, add, leaky ----------
// Tile: 8 rows x 16 cols of output, all 64 channels. Wave w = output row w.

__global__ void __launch_bounds__(256)
conv2_fused(const unsigned* __restrict__ T, const float* __restrict__ x,
            const unsigned* __restrict__ w2hi, const unsigned* __restrict__ w2lo,
            const unsigned* __restrict__ wuhi, const unsigned* __restrict__ wulo,
            const float* __restrict__ bias2, const float* __restrict__ bu,
            float* __restrict__ out) {
    __shared__ uint4 ldsT4[10 * 18 * 16];     // T halo (already hi|lo packed)
    __shared__ uint4 ldsI4[4 * 8 * 16];       // x tile for 1x1 skip, packed
    const unsigned* ldsT = (const unsigned*)ldsT4;
    const unsigned* ldsI = (const unsigned*)ldsI4;
    int bx = blockIdx.x;
    int tx = bx & 15, ty = (bx >> 4) & 31, b = bx >> 9;
    int x0 = tx * 16, y0 = ty * 8;
    int t = threadIdx.x;

    const uint4* Tv = (const uint4*)T;
    for (int e = t; e < 2880; e += 256) {     // 16B-vectorized halo copy
        int c4 = e & 15, hc = (e >> 4) % 18, hr = e / 288;
        int yy = y0 + hr - 1, xx = x0 + hc - 1;
        uint4 v = {0u, 0u, 0u, 0u};
        if (yy >= 0 && yy < 256 && xx >= 0 && xx < 256)
            v = Tv[(((b * 256 + yy) * 256 + xx) << 4) + c4];
        ldsT4[e] = v;
    }
    const float4* xv = (const float4*)x;
    for (int e = t; e < 512; e += 256) {
        int c4 = e & 15, xc = (e >> 4) & 7, xr = e >> 7;
        float4 f = xv[((b * 128 + (y0 >> 1) + xr) * 128 + (x0 >> 1) + xc) * 16 + c4];
        ldsI4[e] = pack_split4(f);
    }
    __syncthreads();

    int lane = t & 31, w = t >> 5;
    int kh = lane >> 4, ml = lane & 15;

    v8f acc[4], accU[4];
#pragma unroll
    for (int ns = 0; ns < 4; ++ns) { acc[ns] = {}; accU[ns] = {}; }

    // conv2 main GEMM (K = 9*64 = 576)
    for (int it = 0; it < 18; ++it) {
        int blk = it >> 1;
        int ky = blk / 3, kx = blk % 3;
        int cc0 = (it & 1) << 5;
        v16bf Ah, Al;
        build_afrags(&ldsT[((w + ky) * 18 + (ml + kx)) * 64 + cc0 + (kh << 3)], Ah, Al);
#pragma unroll
        for (int ns = 0; ns < 4; ++ns) {
            const unsigned* ph = w2hi + (((it * 4 + ns) * 32 + lane) << 3);
            const unsigned* pl = w2lo + (((it * 4 + ns) * 32 + lane) << 3);
            v16bf Bh = load_wfrag(ph);
            v16bf Bl = load_wfrag(pl);
            acc[ns] = wmma_bf16(Ah, Bl, acc[ns]);
            acc[ns] = wmma_bf16(Al, Bh, acc[ns]);
            acc[ns] = wmma_bf16(Ah, Bh, acc[ns]);
        }
    }

    // 1x1 skip branch (K = 64), duplicated x rows give nearest-upsampling
#pragma unroll
    for (int it = 0; it < 2; ++it) {
        int cc0 = it << 5;
        v16bf Ah, Al;
        build_afrags(&ldsI[((w >> 1) * 8 + (ml >> 1)) * 64 + cc0 + (kh << 3)], Ah, Al);
#pragma unroll
        for (int ns = 0; ns < 4; ++ns) {
            const unsigned* ph = wuhi + (((it * 4 + ns) * 32 + lane) << 3);
            const unsigned* pl = wulo + (((it * 4 + ns) * 32 + lane) << 3);
            v16bf Bh = load_wfrag(ph);
            v16bf Bl = load_wfrag(pl);
            accU[ns] = wmma_bf16(Ah, Bl, accU[ns]);
            accU[ns] = wmma_bf16(Al, Bh, accU[ns]);
            accU[ns] = wmma_bf16(Ah, Bh, accU[ns]);
        }
    }

    int oy = y0 + w;
#pragma unroll
    for (int ns = 0; ns < 4; ++ns) {
        int n = ns * 16 + ml;
        float b2v = bias2[n], buv = bu[n];
#pragma unroll
        for (int v = 0; v < 8; ++v) {
            int m = v + (kh << 3);
            int ox = x0 + m;
            float c2 = leaky(acc[ns][v] + b2v);
            float o = leaky(c2 + accU[ns][v] + buv);
            out[(((b * 256 + oy) * 256 + ox) << 6) + n] = o;
        }
    }
}

// ---------- host ----------

extern "C" void kernel_launch(void* const* d_in, const int* in_sizes, int n_in,
                              void* d_out, int out_size, void* d_ws, size_t ws_size,
                              hipStream_t stream) {
    (void)in_sizes; (void)n_in; (void)out_size; (void)ws_size;
    const float* x  = (const float*)d_in[0];
    const float* W1 = (const float*)d_in[1];
    const float* b1 = (const float*)d_in[2];
    const float* g1 = (const float*)d_in[3];
    const float* be1= (const float*)d_in[4];
    const float* m1 = (const float*)d_in[5];
    const float* v1 = (const float*)d_in[6];
    const float* W2 = (const float*)d_in[7];
    const float* b2 = (const float*)d_in[8];
    const float* g2 = (const float*)d_in[9];
    const float* be2= (const float*)d_in[10];
    const float* m2 = (const float*)d_in[11];
    const float* v2 = (const float*)d_in[12];
    const float* Wu = (const float*)d_in[13];
    const float* bu = (const float*)d_in[14];
    float* out = (float*)d_out;

    // workspace layout (bytes); requires ws_size >= ~270 MB
    char* ws = (char*)d_ws;
    float*    s1    = (float*)(ws + 0);          // 256 f
    float*    bias1 = (float*)(ws + 1024);       // 256 f
    float*    s2    = (float*)(ws + 2048);       // 64 f
    float*    bias2 = (float*)(ws + 2304);       // 64 f
    unsigned* w1hi  = (unsigned*)(ws + 4096);    // 18*16*32*8 = 73728 dwords
    unsigned* w1lo  = (unsigned*)(ws + 299008);
    unsigned* w2hi  = (unsigned*)(ws + 593920);  // 18*4*32*8 = 18432 dwords
    unsigned* w2lo  = (unsigned*)(ws + 667648);
    unsigned* wuhi  = (unsigned*)(ws + 741376);  // 2*4*32*8 = 2048 dwords
    unsigned* wulo  = (unsigned*)(ws + 749568);
    unsigned* T     = (unsigned*)(ws + (1u << 20)); // 16*256*256*64 dwords = 256 MB

    fold_bn<<<1, 256, 0, stream>>>(b1, g1, be1, m1, v1, b2, g2, be2, m2, v2,
                                   s1, s2, bias1, bias2);

    pack_weights<<<288, 256, 0, stream>>>(W1, s1, w1hi, w1lo, 256, 16, 73728);
    pack_weights<<< 72, 256, 0, stream>>>(W2, s2, w2hi, w2lo,  64,  4, 18432);
    pack_weights<<<  8, 256, 0, stream>>>(Wu, (const float*)nullptr, wuhi, wulo,
                                          64, 4, 2048);

    // 16 batches * (128/8)^2 tiles = 4096 workgroups
    conv1_subpixel<<<4096, 256, 0, stream>>>(x, w1hi, w1lo, bias1, T);

    // 16 batches * (256/8)*(256/16) tiles = 8192 workgroups
    conv2_fused<<<8192, 256, 0, stream>>>(T, x, w2hi, w2lo, wuhi, wulo,
                                          bias2, bu, out);
}
